// MyTransformerEncoderLayer_89515708383410
// MI455X (gfx1250) — matrix operand order repaired and checked by reading, not compiled, not run on hardware
//
#include <hip/hip_runtime.h>

typedef __bf16 bf16_t;
typedef __attribute__((ext_vector_type(16))) __bf16 v16bf;
typedef __attribute__((ext_vector_type(8)))  __bf16 v8bf;
typedef __attribute__((ext_vector_type(8)))  float  v8f;
typedef __attribute__((ext_vector_type(4)))  unsigned int v4u;
typedef __attribute__((ext_vector_type(8)))  int v8i;
typedef __attribute__((ext_vector_type(4)))  int v4i;

union ABu { v16bf v; struct { v8bf lo, hi; } s; };

#define S_   2048
#define B_   2
#define E_   1024
#define H_   16
#define D_   64
#define DFF_ 4096
#define NTOK (S_*B_)

// ------------------------------------------------------------------
// TDM: issue a 2D tensor-tile load (global -> LDS), tracked by TENSORcnt.
// Descriptor layout per CDNA5 ISA.
// tile_x elements along dim0 (contiguous), tile_y rows, row pitch = stride
// (elements). data_size = 2 bytes (bf16). LDS dest is contiguous.
// Toolchain uses the 6-arg builtin: (g0, g1, g2, g3, extra, cpol).
// ------------------------------------------------------------------
__device__ __forceinline__ void tdm_load_2d_bf16(unsigned int lds_off,
                                                 const void* gptr,
                                                 unsigned int tile_x,
                                                 unsigned int tile_y,
                                                 unsigned long long stride,
                                                 unsigned long long tensor_x,
                                                 unsigned long long tensor_y) {
  const unsigned long long ga = (unsigned long long)(uintptr_t)gptr;
  v4u g0;
  g0[0] = 1u;                                   // count=1 (valid), user mode
  g0[1] = lds_off;                              // lds_addr (bytes)
  g0[2] = (unsigned int)ga;                     // global_addr[31:0]
  g0[3] = (unsigned int)(ga >> 32) | (2u << 30);// global_addr[56:32] | type=2
  v8i g1;
  g1[0] = (int)(1u << 16);                      // wg_mask=0, data_size=1 (2B)
  g1[1] = (int)((tensor_x & 0xFFFFu) << 16);    // tensor_dim0[15:0]
  g1[2] = (int)(((tensor_x >> 16) & 0xFFFFu) | ((tensor_y & 0xFFFFu) << 16));
  g1[3] = (int)(((tensor_y >> 16) & 0xFFFFu) | (tile_x << 16)); // tile_dim0
  g1[4] = (int)tile_y;                          // tile_dim1; tile_dim2=0
  g1[5] = (int)(unsigned int)(stride & 0xFFFFFFFFu);           // dim0 stride lo
  g1[6] = (int)(unsigned int)((stride >> 32) & 0xFFFFu);       // dim0 stride hi
  g1[7] = 0;                                    // dim1 stride hi (unused, 2D)
  v4i g2 = {0, 0, 0, 0};
  v4i g3 = {0, 0, 0, 0};
  v8i g4 = {0, 0, 0, 0, 0, 0, 0, 0};
  __builtin_amdgcn_tensor_load_to_lds(g0, g1, g2, g3, g4, 0);
}

// ------------------------------------------------------------------
// prep: fp32 -> bf16 convert, and transposed convert (W[N,K] -> Wt[K,N])
// ------------------------------------------------------------------
__global__ __launch_bounds__(256) void cvt_bf16_k(const float* __restrict__ in,
                                                  bf16_t* __restrict__ out, int n) {
  int i = blockIdx.x * 256 + threadIdx.x;
  if (i < n) out[i] = (bf16_t)in[i];
}

__global__ __launch_bounds__(256) void transpose_bf16_k(const float* __restrict__ in,
                                                        bf16_t* __restrict__ out,
                                                        int N, int K) {
  int i = blockIdx.x * 256 + threadIdx.x;
  if (i < N * K) {
    int n = i / K;
    int k = i - n * K;
    out[(size_t)k * N + n] = (bf16_t)in[i];
  }
}

// ------------------------------------------------------------------
// GEMM: C[M,N] = A[M,K] (bf16, row-major) x Bt[K,N] (bf16, row-major) + bias
// EPI: 0 = store bf16, 1 = relu + store bf16, 2 = store fp32
// tile 128x128x32, 256 threads (8 waves), wave computes 32x64 (2x4 wmma).
// Double-buffered LDS staging via Tensor Data Mover (wave 0 issues,
// TENSORcnt waited, workgroup barrier publishes).
// ------------------------------------------------------------------
template <int EPI>
__global__ __launch_bounds__(256, 2)
void gemm_bf16_k(const bf16_t* __restrict__ A, const bf16_t* __restrict__ Bt,
                 const float* __restrict__ bias, void* __restrict__ Cout,
                 int M, int N, int K) {
  __shared__ __align__(32) bf16_t As[2][128][32];   // 2 x 8KB
  __shared__ __align__(32) bf16_t Bs[2][32][128];   // 2 x 8KB

  const int tid   = threadIdx.x;
  const int lane  = tid & 31;
  const int wave  = tid >> 5;       // 0..7
  const int wr    = wave >> 1;      // 0..3  -> row group of 32
  const int wc    = wave & 1;       // 0..1  -> col group of 64
  const int l15   = lane & 15;
  const int lhalf = lane >> 4;
  const int m0    = blockIdx.y * 128;
  const int n0    = blockIdx.x * 128;

  v8f acc[2][4] = {};

  const unsigned int ldsA0 = (unsigned int)(uintptr_t)&As[0][0][0];
  const unsigned int ldsA1 = (unsigned int)(uintptr_t)&As[1][0][0];
  const unsigned int ldsB0 = (unsigned int)(uintptr_t)&Bs[0][0][0];
  const unsigned int ldsB1 = (unsigned int)(uintptr_t)&Bs[1][0][0];

  const int nk = K >> 5;

  // prologue: stage slab 0 into buffer 0
  if (wave == 0) {
    tdm_load_2d_bf16(ldsA0, A + (size_t)m0 * K, 32, 128,
                     (unsigned long long)K, (unsigned long long)K,
                     (unsigned long long)M);
    tdm_load_2d_bf16(ldsB0, Bt + n0, 128, 32,
                     (unsigned long long)N, (unsigned long long)N,
                     (unsigned long long)K);
  }

  for (int i = 0; i < nk; ++i) {
    if (wave == 0) __builtin_amdgcn_s_wait_tensorcnt(0);
    __syncthreads();   // buffer (i&1) ready; previous reads of (i+1)&1 done

    const int cur = i & 1;
    if (wave == 0 && i + 1 < nk) {
      const int k1 = (i + 1) << 5;
      tdm_load_2d_bf16(cur ? ldsA0 : ldsA1, A + (size_t)m0 * K + k1, 32, 128,
                       (unsigned long long)K, (unsigned long long)K,
                       (unsigned long long)M);
      tdm_load_2d_bf16(cur ? ldsB0 : ldsB1, Bt + (size_t)k1 * N + n0, 128, 32,
                       (unsigned long long)N, (unsigned long long)N,
                       (unsigned long long)K);
    }

    // A fragments: 16x32 bf16. lane<16: row=l15, K {0..7,16..23}; lane>=16: K {8..15,24..31}
    v16bf af[2];
#pragma unroll
    for (int rb = 0; rb < 2; ++rb) {
      const int row = wr * 32 + rb * 16 + l15;
      const int kk  = lhalf * 8;
      ABu u;
      u.s.lo = *(const v8bf*)&As[cur][row][kk];
      u.s.hi = *(const v8bf*)&As[cur][row][kk + 16];
      af[rb] = u.v;
    }
    // B fragments: 32x16 bf16. col=l15; lane<16: K {0..7,16..23}; lane>=16: K {8..15,24..31}
    v16bf bfm[4];
#pragma unroll
    for (int cb = 0; cb < 4; ++cb) {
      const int col = wc * 64 + cb * 16 + l15;
#pragma unroll
      for (int s = 0; s < 16; ++s)
        bfm[cb][s] = Bs[cur][lhalf * 8 + (s & 7) + ((s >> 3) << 4)][col];
    }
#pragma unroll
    for (int rb = 0; rb < 2; ++rb)
#pragma unroll
      for (int cb = 0; cb < 4; ++cb)
        acc[rb][cb] = __builtin_amdgcn_wmma_f32_16x16x32_bf16(
            false, af[rb], false, bfm[cb], (short)0, acc[rb][cb], false, false);
  }

  // epilogue: C layout -> VGPR v: lanes 0-15 row=v, lanes 16-31 row=v+8; lane&15 = col
#pragma unroll
  for (int rb = 0; rb < 2; ++rb) {
    const int rbase = m0 + wr * 32 + rb * 16 + 8 * lhalf;
#pragma unroll
    for (int cb = 0; cb < 4; ++cb) {
      const int col = n0 + wc * 64 + cb * 16 + l15;
      const float bv = bias[col];
#pragma unroll
      for (int v = 0; v < 8; ++v) {
        const int row = rbase + v;
        float val = acc[rb][cb][v] + bv;
        if (EPI == 1) val = val > 0.f ? val : 0.f;
        if (EPI == 2) ((float*)Cout)[(size_t)row * N + col] = val;
        else          ((bf16_t*)Cout)[(size_t)row * N + col] = (bf16_t)val;
      }
    }
  }
}

// ------------------------------------------------------------------
// Flash attention: per (b,h), 128 query rows per block (8 waves x 16 rows),
// streaming over keys in chunks of 32 with online softmax.
// q/k/v/o layout: [(s*B + b)*E + h*64 + d] (bf16)
// ------------------------------------------------------------------
__global__ __launch_bounds__(256, 2)
void flash_attn_k(const bf16_t* __restrict__ q, const bf16_t* __restrict__ k,
                  const bf16_t* __restrict__ v, bf16_t* __restrict__ o) {
  __shared__ __align__(32) bf16_t Kt[64][32];      // [d][key]
  __shared__ __align__(32) bf16_t Vs[32][64];      // [key][d]
  __shared__ __align__(32) bf16_t Ps[8][16][32];   // per-wave P tile

  const int tid   = threadIdx.x;
  const int lane  = tid & 31;
  const int wave  = tid >> 5;
  const int l15   = lane & 15;
  const int lhalf = lane >> 4;

  const int bh = blockIdx.y;
  const int b  = bh % B_;
  const int h  = bh / B_;
  const int qbase = blockIdx.x * 128;

  const float scale = 0.125f;      // 1/sqrt(64)
  const float LOG2E = 1.4426950408889634f;

  // load Q fragments (16 rows x 64 d = two 16x32 A-fragments), keep in regs
  v16bf aq[2];
  {
    const int qrow = qbase + wave * 16 + l15;
    const size_t qoff = ((size_t)qrow * B_ + b) * E_ + h * 64;
#pragma unroll
    for (int i = 0; i < 2; ++i) {
      const int d0 = i * 32 + lhalf * 8;
      ABu u;
      u.s.lo = *(const v8bf*)(q + qoff + d0);
      u.s.hi = *(const v8bf*)(q + qoff + d0 + 16);
      aq[i] = u.v;
    }
  }

  v8f oacc[4] = {};
  float mrow[8], lrow[8];
#pragma unroll
  for (int i = 0; i < 8; ++i) { mrow[i] = -3.0e38f; lrow[i] = 0.f; }

  const int ldkey = tid >> 3;             // 0..31
  const int lddsg = (tid & 7) * 8;        // 0..56

  for (int kc = 0; kc < S_; kc += 32) {
    // cooperative load of K chunk (transposed into LDS) and V chunk
    {
      const size_t goff = ((size_t)(kc + ldkey) * B_ + b) * E_ + h * 64 + lddsg;
      v8bf kv = *(const v8bf*)(k + goff);
#pragma unroll
      for (int j = 0; j < 8; ++j) Kt[lddsg + j][ldkey] = kv[j];
      *(v8bf*)&Vs[ldkey][lddsg] = *(const v8bf*)(v + goff);
      if (kc + 32 < S_) {
        __builtin_prefetch(k + goff + (size_t)32 * B_ * E_, 0, 1);
        __builtin_prefetch(v + goff + (size_t)32 * B_ * E_, 0, 1);
      }
    }
    __syncthreads();

    // scores: S(16x32) = Q(16x64) x K^T(64x32) -> two 16x16 f32 tiles
    v8f s0 = {}, s1 = {};
#pragma unroll
    for (int i = 0; i < 2; ++i) {
      v16bf bk0, bk1;
#pragma unroll
      for (int s = 0; s < 16; ++s) {
        const int kr = i * 32 + lhalf * 8 + (s & 7) + ((s >> 3) << 4);
        bk0[s] = Kt[kr][l15];
        bk1[s] = Kt[kr][l15 + 16];
      }
      s0 = __builtin_amdgcn_wmma_f32_16x16x32_bf16(false, aq[i], false, bk0, (short)0, s0, false, false);
      s1 = __builtin_amdgcn_wmma_f32_16x16x32_bf16(false, aq[i], false, bk1, (short)0, s1, false, false);
    }

    // online softmax (fully unrolled: no dynamic VGPR indexing)
    float alpha[8];
#pragma unroll
    for (int vv = 0; vv < 8; ++vv) {
      const float a = s0[vv] * scale;
      const float c = s1[vv] * scale;
      float x = fmaxf(a, c);
#pragma unroll
      for (int m = 1; m < 16; m <<= 1) x = fmaxf(x, __shfl_xor(x, m, 32));
      const float mnew = fmaxf(mrow[vv], x);
      alpha[vv] = exp2f((mrow[vv] - mnew) * LOG2E);
      const float p0 = exp2f((a - mnew) * LOG2E);
      const float p1 = exp2f((c - mnew) * LOG2E);
      float ps = p0 + p1;
#pragma unroll
      for (int m = 1; m < 16; m <<= 1) ps += __shfl_xor(ps, m, 32);
      lrow[vv] = lrow[vv] * alpha[vv] + ps;
      mrow[vv] = mnew;
      // stash P (bf16) to per-wave LDS tile: row = vv + 8*lhalf, col = key
      const int prow = vv + 8 * lhalf;
      Ps[wave][prow][l15]      = (bf16_t)p0;
      Ps[wave][prow][l15 + 16] = (bf16_t)p1;
    }
#pragma unroll
    for (int t = 0; t < 4; ++t)
#pragma unroll
      for (int vv = 0; vv < 8; ++vv) oacc[t][vv] *= alpha[vv];

    asm volatile("s_wait_dscnt 0" ::: "memory");   // wave-local LDS RAW

    // reload P as A-fragment (16x32)
    v16bf pA;
    {
      ABu u;
      const int kk = lhalf * 8;
      u.s.lo = *(const v8bf*)&Ps[wave][l15][kk];
      u.s.hi = *(const v8bf*)&Ps[wave][l15][kk + 16];
      pA = u.v;
    }
    // O += P(16x32) x V(32x64)
#pragma unroll
    for (int dc = 0; dc < 4; ++dc) {
      v16bf bv;
#pragma unroll
      for (int s = 0; s < 16; ++s)
        bv[s] = Vs[lhalf * 8 + (s & 7) + ((s >> 3) << 4)][dc * 16 + l15];
      oacc[dc] = __builtin_amdgcn_wmma_f32_16x16x32_bf16(false, pA, false, bv, (short)0, oacc[dc], false, false);
    }
    __syncthreads();
  }

  // normalize and store O (bf16)
  float rinv[8];
#pragma unroll
  for (int vv = 0; vv < 8; ++vv) rinv[vv] = 1.0f / lrow[vv];
#pragma unroll
  for (int dc = 0; dc < 4; ++dc) {
#pragma unroll
    for (int vv = 0; vv < 8; ++vv) {
      const int srow = qbase + wave * 16 + vv + 8 * lhalf;
      const int d    = dc * 16 + l15;
      o[((size_t)srow * B_ + b) * E_ + h * 64 + d] = (bf16_t)(oacc[dc][vv] * rinv[vv]);
    }
  }
}

// ------------------------------------------------------------------
// Residual + LayerNorm over E=1024. One block per token, 256 threads x 4.
// Writes fp32 (xout, if non-null) and bf16 (xbf, if non-null).
// ------------------------------------------------------------------
__global__ __launch_bounds__(256)
void ln_residual_k(const float* __restrict__ a, const float* __restrict__ r,
                   const float* __restrict__ g, const float* __restrict__ be,
                   float* __restrict__ xout, bf16_t* __restrict__ xbf) {
  __shared__ float red[8];
  const int tid = threadIdx.x;
  const size_t base = (size_t)blockIdx.x * E_;

  float vals[4];
  float s = 0.f;
#pragma unroll
  for (int i = 0; i < 4; ++i) {
    const int idx = tid + i * 256;
    const float x = a[base + idx] + r[base + idx];
    vals[i] = x;
    s += x;
  }
#pragma unroll
  for (int m = 16; m >= 1; m >>= 1) s += __shfl_xor(s, m, 32);
  if ((tid & 31) == 0) red[tid >> 5] = s;
  __syncthreads();
  float mean = 0.f;
#pragma unroll
  for (int i = 0; i < 8; ++i) mean += red[i];
  mean *= (1.0f / (float)E_);
  __syncthreads();

  float sv = 0.f;
#pragma unroll
  for (int i = 0; i < 4; ++i) {
    const float d = vals[i] - mean;
    sv += d * d;
  }
#pragma unroll
  for (int m = 16; m >= 1; m >>= 1) sv += __shfl_xor(sv, m, 32);
  if ((tid & 31) == 0) red[tid >> 5] = sv;
  __syncthreads();
  float var = 0.f;
#pragma unroll
  for (int i = 0; i < 8; ++i) var += red[i];
  var *= (1.0f / (float)E_);
  const float rstd = rsqrtf(var + 1e-5f);

#pragma unroll
  for (int i = 0; i < 4; ++i) {
    const int idx = tid + i * 256;
    const float y = (vals[i] - mean) * rstd * g[idx] + be[idx];
    if (xout) xout[base + idx] = y;
    if (xbf)  xbf[base + idx]  = (bf16_t)y;
  }
}

// ------------------------------------------------------------------
// host side
// ------------------------------------------------------------------
extern "C" void kernel_launch(void* const* d_in, const int* in_sizes, int n_in,
                              void* d_out, int out_size, void* d_ws, size_t ws_size,
                              hipStream_t stream) {
  const float* src = (const float*)d_in[0];
  const float* Wq  = (const float*)d_in[1];
  const float* bq  = (const float*)d_in[2];
  const float* Wk  = (const float*)d_in[3];
  const float* bk  = (const float*)d_in[4];
  const float* Wv  = (const float*)d_in[5];
  const float* bv  = (const float*)d_in[6];
  const float* Wo  = (const float*)d_in[7];
  const float* bo  = (const float*)d_in[8];
  const float* W1  = (const float*)d_in[9];
  const float* b1  = (const float*)d_in[10];
  const float* W2  = (const float*)d_in[11];
  const float* b2  = (const float*)d_in[12];
  const float* g1  = (const float*)d_in[13];
  const float* be1 = (const float*)d_in[14];
  const float* g2  = (const float*)d_in[15];
  const float* be2 = (const float*)d_in[16];

  uintptr_t p = (uintptr_t)d_ws;
  auto alloc = [&](size_t bytes) -> void* {
    void* r = (void*)p;
    p += (bytes + 255) & ~(size_t)255;
    return r;
  };

  bf16_t* srcb = (bf16_t*)alloc((size_t)NTOK * E_ * 2);
  bf16_t* WqT  = (bf16_t*)alloc((size_t)E_ * E_ * 2);
  bf16_t* WkT  = (bf16_t*)alloc((size_t)E_ * E_ * 2);
  bf16_t* WvT  = (bf16_t*)alloc((size_t)E_ * E_ * 2);
  bf16_t* WoT  = (bf16_t*)alloc((size_t)E_ * E_ * 2);
  bf16_t* W1T  = (bf16_t*)alloc((size_t)E_ * DFF_ * 2);
  bf16_t* W2T  = (bf16_t*)alloc((size_t)DFF_ * E_ * 2);
  bf16_t* qb   = (bf16_t*)alloc((size_t)NTOK * E_ * 2);
  bf16_t* kb   = (bf16_t*)alloc((size_t)NTOK * E_ * 2);
  bf16_t* vb   = (bf16_t*)alloc((size_t)NTOK * E_ * 2);
  bf16_t* attnb    = (bf16_t*)alloc((size_t)NTOK * E_ * 2);
  float*  attnproj = (float*)alloc((size_t)NTOK * E_ * 4);
  float*  x        = (float*)alloc((size_t)NTOK * E_ * 4);
  bf16_t* xb       = (bf16_t*)alloc((size_t)NTOK * E_ * 2);
  bf16_t* h1       = (bf16_t*)alloc((size_t)NTOK * DFF_ * 2);
  float*  ff2      = (float*)alloc((size_t)NTOK * E_ * 4);

  // prep
  {
    const int n = NTOK * E_;
    cvt_bf16_k<<<(n + 255) / 256, 256, 0, stream>>>(src, srcb, n);
  }
  transpose_bf16_k<<<(E_ * E_ + 255) / 256, 256, 0, stream>>>(Wq, WqT, E_, E_);
  transpose_bf16_k<<<(E_ * E_ + 255) / 256, 256, 0, stream>>>(Wk, WkT, E_, E_);
  transpose_bf16_k<<<(E_ * E_ + 255) / 256, 256, 0, stream>>>(Wv, WvT, E_, E_);
  transpose_bf16_k<<<(E_ * E_ + 255) / 256, 256, 0, stream>>>(Wo, WoT, E_, E_);
  transpose_bf16_k<<<(DFF_ * E_ + 255) / 256, 256, 0, stream>>>(W1, W1T, DFF_, E_);
  transpose_bf16_k<<<(E_ * DFF_ + 255) / 256, 256, 0, stream>>>(W2, W2T, E_, DFF_);

  // QKV projections
  {
    dim3 g(E_ / 128, NTOK / 128);
    gemm_bf16_k<0><<<g, 256, 0, stream>>>(srcb, WqT, bq, qb, NTOK, E_, E_);
    gemm_bf16_k<0><<<g, 256, 0, stream>>>(srcb, WkT, bk, kb, NTOK, E_, E_);
    gemm_bf16_k<0><<<g, 256, 0, stream>>>(srcb, WvT, bv, vb, NTOK, E_, E_);
  }

  // attention
  flash_attn_k<<<dim3(S_ / 128, B_ * H_), 256, 0, stream>>>(qb, kb, vb, attnb);

  // output projection (fp32 out)
  gemm_bf16_k<2><<<dim3(E_ / 128, NTOK / 128), 256, 0, stream>>>(attnb, WoT, bo, attnproj, NTOK, E_, E_);

  // LN1: x = LN(src + attn_out)
  ln_residual_k<<<NTOK, 256, 0, stream>>>(src, attnproj, g1, be1, x, xb);

  // FFN
  gemm_bf16_k<1><<<dim3(DFF_ / 128, NTOK / 128), 256, 0, stream>>>(xb, W1T, b1, h1, NTOK, DFF_, E_);
  gemm_bf16_k<2><<<dim3(E_ / 128, NTOK / 128), 256, 0, stream>>>(h1, W2T, b2, ff2, NTOK, E_, DFF_);

  // LN2 -> d_out (fp32)
  ln_residual_k<<<NTOK, 256, 0, stream>>>(x, ff2, g2, be2, (float*)d_out, nullptr);
}
